// SelfAttentionLM_43250320670753
// MI455X (gfx1250) — matrix operand order repaired and checked by reading, not compiled, not run on hardware
//
#include <hip/hip_runtime.h>
#include <hip/hip_bf16.h>

// ---------------------------------------------------------------------------
// CDNA5 (gfx1250) self-attention LM forward.
// Pipeline: embed(+split) -> qkv GEMM -> vT transpose -> flash attention
//           -> proj GEMM -> fc GEMM
// GEMM staging now uses gfx1250 GLOBAL_LOAD_ASYNC_TO_LDS_B128 (ASYNCcnt) with
// double-buffered LDS: async fills of tile k+1 overlap WMMA on tile k.
// ---------------------------------------------------------------------------

typedef __attribute__((ext_vector_type(16))) __bf16 v16bf;
typedef __attribute__((ext_vector_type(4)))  __bf16 v4bf;
typedef __attribute__((ext_vector_type(8)))  float  v8f;

static constexpr int BB   = 2;
static constexpr int TT   = 2048;
static constexpr int EE   = 1024;
static constexpr int HH   = 16;
static constexpr int HD   = 64;
static constexpr int MROW = BB * TT;          // 4096

__device__ __forceinline__ __bf16 f2bf(float f) {
    unsigned u = __float_as_uint(f);
    unsigned r = u + 0x7FFFu + ((u >> 16) & 1u);   // RNE
    return __builtin_bit_cast(__bf16, (unsigned short)(r >> 16));
}
__device__ __forceinline__ float bf2f(__bf16 b) {
    return __uint_as_float(((unsigned)__builtin_bit_cast(unsigned short, b)) << 16);
}
__device__ __forceinline__ v8f v8f_zero() {
    v8f z;
#pragma unroll
    for (int i = 0; i < 8; ++i) z[i] = 0.0f;
    return z;
}
struct U32B { uint4 a, b; };
__device__ __forceinline__ v16bf make_frag(uint4 a, uint4 b) {
    U32B u; u.a = a; u.b = b;
    return __builtin_bit_cast(v16bf, u);
}

// gfx1250 async memory->LDS copy (16B per lane), tracked by ASYNCcnt.
// LDS dest = low 32 bits of generic pointer (wave-relative LDS offset).
__device__ __forceinline__ void async_ld16(void* lds, const void* gptr) {
    unsigned ldsoff = (unsigned)(size_t)lds;
    asm volatile("global_load_async_to_lds_b128 %0, %1, off"
                 :: "v"(ldsoff), "v"(gptr) : "memory");
}
__device__ __forceinline__ void wait_async0() {
    asm volatile("s_wait_asynccnt 0" ::: "memory");
}

// ---------------------------------------------------------------------------
// Embedding + hi/lo split: h = tok_emb[x] + pos_emb  ->  hHi,hLo bf16 [M,E]
// ---------------------------------------------------------------------------
__global__ __launch_bounds__(256) void embed_split_kernel(
    const int* __restrict__ x, const float* __restrict__ tok,
    const float* __restrict__ pos, __bf16* __restrict__ hHi,
    __bf16* __restrict__ hLo)
{
    size_t i  = (size_t)blockIdx.x * blockDim.x + threadIdx.x;
    int    e4 = (int)(i % (EE / 4)) * 4;
    int    bt = (int)(i / (EE / 4));
    int    t  = bt % TT;
    int    tk = x[bt];
    float4 a  = *(const float4*)(tok + (size_t)tk * EE + e4);
    float4 p  = *(const float4*)(pos + (size_t)t  * EE + e4);
    float  s[4] = {a.x + p.x, a.y + p.y, a.z + p.z, a.w + p.w};
    v4bf hi, lo;
#pragma unroll
    for (int j = 0; j < 4; ++j) {
        __bf16 h = f2bf(s[j]);
        hi[j] = h;
        lo[j] = f2bf(s[j] - bf2f(h));
    }
    *(v4bf*)(hHi + (size_t)bt * EE + e4) = hi;
    *(v4bf*)(hLo + (size_t)bt * EE + e4) = lo;
}

// ---------------------------------------------------------------------------
// Weight transpose + split: W f32 [K,N] -> WtHi,WtLo bf16 [N,K]
// ---------------------------------------------------------------------------
__global__ __launch_bounds__(256) void wtrans_split_kernel(
    const float* __restrict__ W, __bf16* __restrict__ WtHi,
    __bf16* __restrict__ WtLo, int K, int N)
{
    __shared__ float tw[32][36];
    const int k0 = blockIdx.x * 32, n0 = blockIdx.y * 32;
    const int tid = threadIdx.x;
    {
        int r = tid >> 3, c = (tid & 7) * 4;
        *(float4*)&tw[r][c] = *(const float4*)(W + (size_t)(k0 + r) * N + n0 + c);
    }
    __syncthreads();
    {
        int n = tid >> 3, kq = (tid & 7) * 4;
        v4bf hi, lo;
#pragma unroll
        for (int i = 0; i < 4; ++i) {
            float x = tw[kq + i][n];
            __bf16 h = f2bf(x);
            hi[i] = h;
            lo[i] = f2bf(x - bf2f(h));
        }
        *(v4bf*)(WtHi + (size_t)(n0 + n) * K + k0 + kq) = hi;
        *(v4bf*)(WtLo + (size_t)(n0 + n) * K + k0 + kq) = lo;
    }
}

// ---------------------------------------------------------------------------
// V transpose (bf16): qkv v-part [T,64] -> vt [64,T] per (b,h), hi and lo
// ---------------------------------------------------------------------------
__global__ __launch_bounds__(256) void vtrans_kernel(
    const __bf16* __restrict__ qkvHi, const __bf16* __restrict__ qkvLo,
    __bf16* __restrict__ vtHi, __bf16* __restrict__ vtLo)
{
    __shared__ __bf16 th[32][36], tl[32][36];
    const int bh = blockIdx.z, b = bh >> 4, h = bh & 15;
    const int t0 = blockIdx.x * 32, d0 = blockIdx.y * 32;
    const int tid = threadIdx.x;
    {
        int r = tid >> 3, c = (tid & 7) * 4;
        size_t src = (size_t)(b * TT + t0 + r) * (3 * EE) + 2 * EE + h * HD + d0 + c;
        *(v4bf*)&th[r][c] = *(const v4bf*)(qkvHi + src);
        *(v4bf*)&tl[r][c] = *(const v4bf*)(qkvLo + src);
    }
    __syncthreads();
    {
        int d = tid >> 3, tq = (tid & 7) * 4;
        v4bf oh, ol;
#pragma unroll
        for (int i = 0; i < 4; ++i) { oh[i] = th[tq + i][d]; ol[i] = tl[tq + i][d]; }
        size_t dst = (size_t)(bh * HD + d0 + d) * TT + t0 + tq;
        *(v4bf*)(vtHi + dst) = oh;
        *(v4bf*)(vtLo + dst) = ol;
    }
}

// ---------------------------------------------------------------------------
// GEMM: C[M,N] = A @ B + bias, bf16 hi/lo inputs (A row-major, B transposed
// [N,K]), fp32 WMMA accumulation, 3-product hi/lo correction.
// Double-buffered LDS filled by GLOBAL_LOAD_ASYNC_TO_LDS_B128; two barriers
// per K-step; async fills of tile k+1 overlap WMMAs on tile k.
// OUT_MODE 0: fp32 C.  OUT_MODE 2: bf16 hi/lo split outputs.
// Block 256 = 8 waves (2M x 4N); WG tile 64x128; wave tile 32x32.
// ---------------------------------------------------------------------------
template <int OUT_MODE>
__global__ __launch_bounds__(256) void gemm_bf16x3(
    const __bf16* __restrict__ AHi, const __bf16* __restrict__ ALo,
    const __bf16* __restrict__ BtHi, const __bf16* __restrict__ BtLo,
    const float* __restrict__ bias, float* __restrict__ C,
    __bf16* __restrict__ CHi, __bf16* __restrict__ CLo,
    int M, int N, int K)
{
    __shared__ __bf16 Ah_s[2][64][40],  Al_s[2][64][40];   // row stride 80B
    __shared__ __bf16 Bh_s[2][128][40], Bl_s[2][128][40];

    const int tid  = threadIdx.x;
    const int wave = tid >> 5, lane = tid & 31;
    const int wm   = wave >> 2, wn = wave & 3;
    const int m0   = blockIdx.y * 64, n0 = blockIdx.x * 128;
    const int half = lane >> 4, l16 = lane & 15;
    const int kA   = half * 8;

    // staging indices (fixed per thread)
    const int ra = tid >> 2, ca = (tid & 3) * 8;       // A: 64 rows x 32 bf16

    v8f acc[2][2];
#pragma unroll
    for (int mt = 0; mt < 2; ++mt)
#pragma unroll
        for (int nt = 0; nt < 2; ++nt) acc[mt][nt] = v8f_zero();

    // async-fill one K-tile (hi+lo of A and B) into LDS buffer bf
    auto stage = [&](int bf, int kk) {
        size_t sA = (size_t)(m0 + ra) * K + kk + ca;
        async_ld16(&Ah_s[bf][ra][ca], AHi + sA);
        async_ld16(&Al_s[bf][ra][ca], ALo + sA);
#pragma unroll
        for (int i = 0; i < 2; ++i) {
            int v = tid + i * 256;
            int r = v >> 2, c = (v & 3) * 8;
            size_t sB = (size_t)(n0 + r) * K + kk + c;
            async_ld16(&Bh_s[bf][r][c], BtHi + sB);
            async_ld16(&Bl_s[bf][r][c], BtLo + sB);
        }
    };

    const int NK = K >> 5;
    stage(0, 0);
    for (int kt = 0; kt < NK; ++kt) {
        const int bf = kt & 1;
        wait_async0();        // my fills of buffer bf are done
        __syncthreads();      // everyone's fills done; prev compute finished
        if (kt + 1 < NK) stage(bf ^ 1, (kt + 1) * 32);
        if (kt + 2 < NK) {    // warm L2 for the tile after next
            __builtin_prefetch(AHi  + (size_t)(m0 + ra) * K + (kt + 2) * 32 + ca, 0, 3);
            __builtin_prefetch(BtHi + (size_t)(n0 + ra) * K + (kt + 2) * 32 + ca, 0, 3);
        }

        // fragments: pure b128 LDS loads, ISA layouts
        v16bf ah[2], al[2], bh[2], bl[2];
#pragma unroll
        for (int mt = 0; mt < 2; ++mt) {
            int row = wm * 32 + mt * 16 + l16;
            ah[mt] = make_frag(*(const uint4*)&Ah_s[bf][row][kA],
                               *(const uint4*)&Ah_s[bf][row][kA + 16]);
            al[mt] = make_frag(*(const uint4*)&Al_s[bf][row][kA],
                               *(const uint4*)&Al_s[bf][row][kA + 16]);
        }
#pragma unroll
        for (int nt = 0; nt < 2; ++nt) {
            int col = wn * 32 + nt * 16 + l16;
            bh[nt] = make_frag(*(const uint4*)&Bh_s[bf][col][kA],
                               *(const uint4*)&Bh_s[bf][col][kA + 16]);
            bl[nt] = make_frag(*(const uint4*)&Bl_s[bf][col][kA],
                               *(const uint4*)&Bl_s[bf][col][kA + 16]);
        }
#pragma unroll
        for (int mt = 0; mt < 2; ++mt)
#pragma unroll
            for (int nt = 0; nt < 2; ++nt) {
                acc[mt][nt] = __builtin_amdgcn_wmma_f32_16x16x32_bf16(
                    false, ah[mt], false, bh[nt], (short)0, acc[mt][nt], false, false);
                acc[mt][nt] = __builtin_amdgcn_wmma_f32_16x16x32_bf16(
                    false, ah[mt], false, bl[nt], (short)0, acc[mt][nt], false, false);
                acc[mt][nt] = __builtin_amdgcn_wmma_f32_16x16x32_bf16(
                    false, al[mt], false, bh[nt], (short)0, acc[mt][nt], false, false);
            }
        __syncthreads();      // done reading buffer bf (refilled next iter)
    }

    // epilogue (C/D layout: VGPR r -> M = r + 8*half, N = lane&15)
#pragma unroll
    for (int mt = 0; mt < 2; ++mt)
#pragma unroll
        for (int nt = 0; nt < 2; ++nt) {
            int n = n0 + wn * 32 + nt * 16 + l16;
            float bv = bias[n];
#pragma unroll
            for (int r = 0; r < 8; ++r) {
                int m = m0 + wm * 32 + mt * 16 + r + half * 8;
                float o = acc[mt][nt][r] + bv;
                if (OUT_MODE == 0) {
                    C[(size_t)m * N + n] = o;
                } else {
                    __bf16 h = f2bf(o);
                    CHi[(size_t)m * N + n] = h;
                    CLo[(size_t)m * N + n] = f2bf(o - bf2f(h));
                }
            }
        }
}

// ---------------------------------------------------------------------------
// Flash attention (bf16 operands straight from global, no LDS staging except
// the P re-layout). Block 128 = 4 independent waves; 16 query rows per wave.
// q,k: bf16(hi); v: hi+lo (2 WMMAs). Softmax scale folded post-WMMA.
// ---------------------------------------------------------------------------
__global__ __launch_bounds__(128) void flash_attn_kernel(
    const __bf16* __restrict__ qkvHi,   // [M, 3E]
    const __bf16* __restrict__ vtHi,    // [BH, 64, T]
    const __bf16* __restrict__ vtLo,
    __bf16* __restrict__ oHi, __bf16* __restrict__ oLo)  // [M, E]
{
    __shared__ __bf16 Pl[4][16][40];

    const int bh = blockIdx.y, b = bh >> 4, h = bh & 15;
    const int q0 = blockIdx.x * 64;
    const int tid = threadIdx.x, wave = tid >> 5, lane = tid & 31;
    const int half = lane >> 4, l16 = lane & 15;
    const int kA = half * 8;
    const int qbase = q0 + wave * 16;
    const int rs = 3 * EE;

    // q fragments (K-dim 64 in two chunks of 32)
    v16bf qf[2];
    {
        const __bf16* qr = qkvHi + (size_t)(b * TT + qbase + l16) * rs + h * HD;
#pragma unroll
        for (int c = 0; c < 2; ++c)
            qf[c] = make_frag(*(const uint4*)(qr + c * 32 + kA),
                              *(const uint4*)(qr + c * 32 + kA + 16));
    }

    v8f acc[4];
#pragma unroll
    for (int nc = 0; nc < 4; ++nc) acc[nc] = v8f_zero();
    float mst[8], lst[8];
#pragma unroll
    for (int r = 0; r < 8; ++r) { mst[r] = -1e30f; lst[r] = 0.0f; }

    const int ntiles = (qbase + 47) >> 5;     // keys <= qbase+15
    for (int j = 0; j < ntiles; ++j) {
        const int kb = j * 32;

        // S = (q k^T)/8 : two 16x16 tiles
        v8f S[2];
#pragma unroll
        for (int st = 0; st < 2; ++st) {
            S[st] = v8f_zero();
            const __bf16* kr =
                qkvHi + (size_t)(b * TT + kb + st * 16 + l16) * rs + EE + h * HD;
#pragma unroll
            for (int c = 0; c < 2; ++c) {
                v16bf bf = make_frag(*(const uint4*)(kr + c * 32 + kA),
                                     *(const uint4*)(kr + c * 32 + kA + 16));
                S[st] = __builtin_amdgcn_wmma_f32_16x16x32_bf16(
                    false, qf[c], false, bf, (short)0, S[st], false, false);
            }
        }
        // scale + causal mask
#pragma unroll
        for (int st = 0; st < 2; ++st)
#pragma unroll
            for (int r = 0; r < 8; ++r) {
                int qi  = qbase + r + half * 8;
                int key = kb + st * 16 + l16;
                S[st][r] = (key > qi) ? -1e30f : S[st][r] * 0.125f;
            }
        // online softmax (row = 16-lane half-wave group)
        float mnew[8], scl[8];
        v8f P[2];
#pragma unroll
        for (int r = 0; r < 8; ++r) {
            float v = fmaxf(S[0][r], S[1][r]);
#pragma unroll
            for (int m = 1; m <= 8; m <<= 1) v = fmaxf(v, __shfl_xor(v, m, 32));
            mnew[r] = fmaxf(mst[r], v);
            scl[r]  = __expf(mst[r] - mnew[r]);
        }
#pragma unroll
        for (int st = 0; st < 2; ++st)
#pragma unroll
            for (int r = 0; r < 8; ++r)
                P[st][r] = __expf(S[st][r] - mnew[r]);
#pragma unroll
        for (int r = 0; r < 8; ++r) {
            float s = P[0][r] + P[1][r];
#pragma unroll
            for (int m = 1; m <= 8; m <<= 1) s += __shfl_xor(s, m, 32);
            lst[r] = lst[r] * scl[r] + s;
            mst[r] = mnew[r];
        }
#pragma unroll
        for (int nc = 0; nc < 4; ++nc)
#pragma unroll
            for (int r = 0; r < 8; ++r) acc[nc][r] *= scl[r];

        // P: C-layout -> LDS (bf16) -> A-fragment
#pragma unroll
        for (int st = 0; st < 2; ++st)
#pragma unroll
            for (int r = 0; r < 8; ++r)
                Pl[wave][r + half * 8][st * 16 + l16] = f2bf(P[st][r]);
        // same-wave DS ops are in-order (ISA ch.11)
        v16bf pf = make_frag(*(const uint4*)&Pl[wave][l16][kA],
                             *(const uint4*)&Pl[wave][l16][kA + 16]);

        // acc += P @ V (hi + lo), V^T pre-transposed: contiguous key chunks
#pragma unroll
        for (int nc = 0; nc < 4; ++nc) {
            size_t vs = (size_t)(bh * HD + nc * 16 + l16) * TT + kb + kA;
            v16bf vfh = make_frag(*(const uint4*)(vtHi + vs),
                                  *(const uint4*)(vtHi + vs + 16));
            v16bf vfl = make_frag(*(const uint4*)(vtLo + vs),
                                  *(const uint4*)(vtLo + vs + 16));
            acc[nc] = __builtin_amdgcn_wmma_f32_16x16x32_bf16(
                false, pf, false, vfh, (short)0, acc[nc], false, false);
            acc[nc] = __builtin_amdgcn_wmma_f32_16x16x32_bf16(
                false, pf, false, vfl, (short)0, acc[nc], false, false);
        }
    }

    // epilogue: split bf16 output (feeds proj GEMM)
#pragma unroll
    for (int nc = 0; nc < 4; ++nc)
#pragma unroll
        for (int r = 0; r < 8; ++r) {
            int qi = qbase + r + half * 8;
            float o = acc[nc][r] / lst[r];
            size_t idx = (size_t)(b * TT + qi) * EE + h * HD + nc * 16 + l16;
            __bf16 hb = f2bf(o);
            oHi[idx] = hb;
            oLo[idx] = f2bf(o - bf2f(hb));
        }
}

// ---------------------------------------------------------------------------
// Launch. Workspace layout (bytes), with fc_w^T aliasing dead early buffers:
//   [0)  projWtHi 2MB | projWtLo 2MB | projHi 8MB | projLo 8MB   (live late)
//   [20MB) hHi 8 | hLo 8 | qkvWtHi 6 | qkvWtLo 6 | qkvHi 24 | qkvLo 24
//          | vtHi 8 | vtLo 8 | attnHi 8 | attnLo 8              (dead by fc)
//   fcWtHi 64MB @20MB, fcWtLo 64MB @84MB  -> peak 148MB
// ---------------------------------------------------------------------------
extern "C" void kernel_launch(void* const* d_in, const int* in_sizes, int n_in,
                              void* d_out, int out_size, void* d_ws, size_t ws_size,
                              hipStream_t stream)
{
    (void)in_sizes; (void)n_in; (void)out_size; (void)ws_size;

    const int*   x       = (const int*)  d_in[0];
    const float* tok_emb = (const float*)d_in[1];
    const float* pos_emb = (const float*)d_in[2];
    const float* qkv_w   = (const float*)d_in[3];
    const float* qkv_b   = (const float*)d_in[4];
    const float* proj_w  = (const float*)d_in[5];
    const float* proj_b  = (const float*)d_in[6];
    const float* fc_w    = (const float*)d_in[7];
    const float* fc_b    = (const float*)d_in[8];
    float*       outp    = (float*)d_out;

    char* w = (char*)d_ws;
    const size_t MB = 1u << 20;
    __bf16* projWtHi = (__bf16*)(w + 0 * MB);
    __bf16* projWtLo = (__bf16*)(w + 2 * MB);
    __bf16* projHi   = (__bf16*)(w + 4 * MB);
    __bf16* projLo   = (__bf16*)(w + 12 * MB);
    __bf16* hHi      = (__bf16*)(w + 20 * MB);
    __bf16* hLo      = (__bf16*)(w + 28 * MB);
    __bf16* qkvWtHi  = (__bf16*)(w + 36 * MB);
    __bf16* qkvWtLo  = (__bf16*)(w + 42 * MB);
    __bf16* qkvHi    = (__bf16*)(w + 48 * MB);
    __bf16* qkvLo    = (__bf16*)(w + 72 * MB);
    __bf16* vtHi     = (__bf16*)(w + 96 * MB);
    __bf16* vtLo     = (__bf16*)(w + 104 * MB);
    __bf16* attnHi   = (__bf16*)(w + 112 * MB);
    __bf16* attnLo   = (__bf16*)(w + 120 * MB);
    __bf16* fcWtHi   = (__bf16*)(w + 20 * MB);   // aliases h/qkvWt (dead by then)
    __bf16* fcWtLo   = (__bf16*)(w + 84 * MB);   // aliases qkv/vt/attn (dead)

    // 1) embedding + split
    embed_split_kernel<<<dim3(MROW * EE / 4 / 256), dim3(256), 0, stream>>>(
        x, tok_emb, pos_emb, hHi, hLo);
    // 2) qkv_w^T split
    wtrans_split_kernel<<<dim3(EE / 32, 3 * EE / 32), dim3(256), 0, stream>>>(
        qkv_w, qkvWtHi, qkvWtLo, EE, 3 * EE);
    // 3) qkv GEMM (split bf16 out)
    gemm_bf16x3<2><<<dim3(3 * EE / 128, MROW / 64), dim3(256), 0, stream>>>(
        hHi, hLo, qkvWtHi, qkvWtLo, qkv_b, nullptr, qkvHi, qkvLo,
        MROW, 3 * EE, EE);
    // 4) V transpose (hi+lo)
    vtrans_kernel<<<dim3(TT / 32, HD / 32, BB * HH), dim3(256), 0, stream>>>(
        qkvHi, qkvLo, vtHi, vtLo);
    // 5) flash attention (split bf16 out)
    flash_attn_kernel<<<dim3(TT / 64, BB * HH), dim3(128), 0, stream>>>(
        qkvHi, vtHi, vtLo, attnHi, attnLo);
    // 6) proj_w^T split
    wtrans_split_kernel<<<dim3(EE / 32, EE / 32), dim3(256), 0, stream>>>(
        proj_w, projWtHi, projWtLo, EE, EE);
    // 7) proj GEMM (split bf16 out)
    gemm_bf16x3<2><<<dim3(EE / 128, MROW / 64), dim3(256), 0, stream>>>(
        attnHi, attnLo, projWtHi, projWtLo, proj_b, nullptr, projHi, projLo,
        MROW, EE, EE);
    // 8) fc_w^T split (aliased region is dead now)
    wtrans_split_kernel<<<dim3(EE / 32, 32000 / 32), dim3(256), 0, stream>>>(
        fc_w, fcWtHi, fcWtLo, EE, 32000);
    // 9) fc GEMM -> fp32 logits
    gemm_bf16x3<0><<<dim3(32000 / 128, MROW / 64), dim3(256), 0, stream>>>(
        projHi, projLo, fcWtHi, fcWtLo, fc_b, outp, nullptr, nullptr,
        MROW, 32000, EE);
}
